// BigbirdBlockSpareAttention_47399259079377
// MI455X (gfx1250) — matrix operand order repaired and checked by reading, not compiled, not run on hardware
//
#include <hip/hip_runtime.h>
#include <hip/hip_bf16.h>

// ---------------------------------------------------------------------------
// BigBird block-sparse attention for MI455X (gfx1250, wave32, WMMA + TDM).
// B=2, H=16, SEQ=4096, D=64, block=64, R=3 random blocks.
// One workgroup (128 thr = 4 waves) per (b, h, row-block); flash-attention
// over the row-block's key-block list with v_wmma_f32_16x16x32_f16.
// K/V tiles are DMA'd by the Tensor Data Mover into a double-buffered f32
// LDS staging area (TENSORcnt-pipelined), then converted to f16 in LDS.
// Softmax row-sums are computed by WMMA against an all-ones B fragment.
// ---------------------------------------------------------------------------

typedef __attribute__((ext_vector_type(16))) _Float16     v16h;
typedef __attribute__((ext_vector_type(8)))  _Float16     v8h;
typedef __attribute__((ext_vector_type(8)))  float        v8f;
typedef __attribute__((ext_vector_type(4)))  unsigned int u32x4;
typedef __attribute__((ext_vector_type(8)))  int          i32x8;
typedef __attribute__((ext_vector_type(4)))  int          i32x4;

#define NB    64
#define BLK   64
#define DH    64
#define LSTR  72          // LDS f16 row stride (144B = 36 dwords: conflict-free)
#define NEGV  -10000.0f
#define BSZ   2
#define HN    16
#define SEQL  4096

#if __has_include(<hip/amd_detail/amd_gfx1250_TDM.h>)
#define TDM_6ARG 1        // therock-10.0 headers -> clang-23 6-arg builtin
#endif

// ---- Tensor Data Mover: 1-D tile of 4096 dwords (16KB) global -> LDS ------
// D# group0: count=1, lds_addr[63:32], global_addr[120:64], type=2 [127:126]
// D# group1: data_size=4B [17:16], tensor_dim0=4096 [79:48], tensor_dim1=1,
//            tile_dim0=4096 [127:112], tile_dim1/2 unused, dim0_stride=4096
#if __has_builtin(__builtin_amdgcn_tensor_load_to_lds)
#define HAVE_TDM 1
__device__ __forceinline__ void tdm_load_16kb(const float* gsrc, float* ldst) {
  unsigned long long ga = (unsigned long long)(uintptr_t)gsrc;
  u32x4 g0;
  g0[0] = 1u;                                   // count=1, is_restore=0
  g0[1] = (unsigned)(uintptr_t)ldst;            // LDS byte address
  g0[2] = (unsigned)ga;                         // global_addr[31:0]
  g0[3] = (unsigned)((ga >> 32) & 0x01FFFFFFu) | 0x80000000u; // [56:32]|type=2
  i32x8 g1;
  g1[0] = 0x00020000;   // workgroup_mask=0, data_size=2 (4B)
  g1[1] = 0x10000000;   // tensor_dim0 lo16 = 4096 (bits 63:48)
  g1[2] = 0x00010000;   // tensor_dim0 hi = 0, tensor_dim1 lo16 = 1
  g1[3] = 0x10000000;   // tensor_dim1 hi = 0, tile_dim0 = 4096
  g1[4] = 0;            // tile_dim1 = 0 (unused), tile_dim2 = 0
  g1[5] = 0x00001000;   // tensor_dim0_stride = 4096
  g1[6] = 0; g1[7] = 0;
  i32x4 z4 = {0, 0, 0, 0};
#ifdef TDM_6ARG
  i32x8 z8 = {0, 0, 0, 0, 0, 0, 0, 0};
  __builtin_amdgcn_tensor_load_to_lds(g0, g1, z4, z4, z8, 0);
#else
  __builtin_amdgcn_tensor_load_to_lds(g0, g1, z4, z4, 0);
#endif
}
#endif

// ---- half-wave (16-lane) max reduction ------------------------------------
#if __has_builtin(__builtin_amdgcn_permlane16)
__device__ __forceinline__ float pl16(float x, unsigned s0, unsigned s1) {
  int xi = __float_as_int(x);
  return __int_as_float(__builtin_amdgcn_permlane16(xi, xi, s0, s1, false, false));
}
__device__ __forceinline__ float rowmax16(float x) {
  x = fmaxf(x, pl16(x, 0x67452301u, 0xEFCDAB89u));   // xor 1
  x = fmaxf(x, pl16(x, 0x45670123u, 0xCDEF89ABu));   // xor 3
  x = fmaxf(x, pl16(x, 0x32107654u, 0xBA98FEDCu));   // xor 4
  x = fmaxf(x, pl16(x, 0xFEDCBA98u, 0x76543210u));   // xor 8
  return x;
}
#else
__device__ __forceinline__ float rowmax16(float x) {
#pragma unroll
  for (int s = 1; s < 16; s <<= 1) x = fmaxf(x, __shfl_xor(x, s, 32));
  return x;
}
#endif

// A-fragment (16x32 f16, ISA 7.12.2): lane<16 -> M=lane, K {c+h*8..+7, c+16+h*8..+7}
__device__ __forceinline__ v16h load_a_frag(const _Float16* base, int row, int col, int half) {
  const _Float16* p = base + row * LSTR + col + half * 8;
  v16h a;
  ((v8h*)&a)[0] = *(const v8h*)(p);
  ((v8h*)&a)[1] = *(const v8h*)(p + 16);
  return a;
}
// B-fragment (32x16 f16): lane n = lane&15 holds K {col+half*8..+7, col+16+half*8..+7}
__device__ __forceinline__ v16h load_b_frag(const _Float16* base, int row, int col, int half) {
  const _Float16* p = base + row * LSTR + col + half * 8;
  v16h bb;
  ((v8h*)&bb)[0] = *(const v8h*)(p);
  ((v8h*)&bb)[1] = *(const v8h*)(p + 16);
  return bb;
}

// Key-block id for iteration `it` of row-block `rb` (BigBird block list).
__device__ __forceinline__ int kb_of(int it, int rb, int r0, int r1, int r2) {
  if (rb == 0 || rb == NB - 1) return it;                 // full rows
  if (rb == 1) {                                          // {0,1,2,63, r0,r1,r2}
    if (it < 3)  return it;
    if (it == 3) return NB - 1;
    return (it == 4) ? r0 : ((it == 5) ? r1 : r2);
  }
  if (rb == NB - 2) {                                     // {0,61,62,63, r0,r1,r2}
    if (it == 0) return 0;
    if (it < 4)  return NB - 4 + it;
    return (it == 4) ? r0 : ((it == 5) ? r1 : r2);
  }
  if (it == 0) return 0;                                  // {0, rb-1..rb+1, 63, r*}
  if (it < 4)  return rb - 2 + it;
  if (it == 4) return NB - 1;
  return (it == 5) ? r0 : ((it == 6) ? r1 : r2);
}

__global__ __launch_bounds__(128)
void bigbird_sparse_attn_kernel(const float* __restrict__ Q,
                                const float* __restrict__ K,
                                const float* __restrict__ V,
                                const int*   __restrict__ rand_attn,
                                const float* __restrict__ from_mask,
                                const float* __restrict__ to_mask,
                                float*       __restrict__ out)
{
  __shared__ __align__(16) _Float16 sQ [BLK * LSTR];     // Q  f16 row-major
  __shared__ __align__(16) _Float16 sKh[BLK * LSTR];     // K  f16 [key][d]
  __shared__ __align__(16) _Float16 sVt[DH  * LSTR];     // V  f16 [d][key]
  __shared__ __align__(16) _Float16 sP [4][16 * LSTR];   // per-wave P strip
#ifdef HAVE_TDM
  __shared__ __align__(16) float Kf32[2][BLK * DH];      // TDM staging (double buf)
  __shared__ __align__(16) float Vf32[2][BLK * DH];
#endif

  const int tid  = threadIdx.x;
  const int lane = tid & 31;
  const int wq   = tid >> 5;
  const int half = lane >> 4;
  const int l16  = lane & 15;

  const int rb = blockIdx.x & (NB - 1);
  const int h  = (blockIdx.x >> 6) & (HN - 1);
  const int b  = blockIdx.x >> 10;

  const size_t bh  = (size_t)(b * HN + h);
  const float* Qb  = Q + (bh * SEQL + (size_t)rb * BLK) * DH;
  const float* Kb  = K + bh * SEQL * DH;
  const float* Vb  = V + bh * SEQL * DH;
  const float* tmb = to_mask + (size_t)b * SEQL;

  int r0 = 0, r1 = 0, r2 = 0;
  if (rb >= 1 && rb <= NB - 2) {
    const int* rp = rand_attn + (bh * (NB - 2) + (size_t)(rb - 1)) * 3;
    r0 = rp[0]; r1 = rp[1]; r2 = rp[2];
  }
  const bool full   = (rb == 0 || rb == NB - 1);
  const bool edge   = (rb == 1 || rb == NB - 2);
  const int  nkb    = full ? NB : (edge ? 7 : 8);
  const int  rstart = full ? NB : (edge ? 4 : 5);   // it >= rstart: random block

#ifdef HAVE_TDM
  // prologue: DMA key block 0 into staging buffer 0 (2 tensor ops, wave 0 only)
  {
    const int kb0 = kb_of(0, rb, r0, r1, r2);
    if (wq == 0) {
      tdm_load_16kb(Kb + (size_t)kb0 * BLK * DH, Kf32[0]);
      tdm_load_16kb(Vb + (size_t)kb0 * BLK * DH, Vf32[0]);
    }
  }
#endif

  // ---- stage Q (f32 -> f16) and build the two constant A-fragments --------
  {
    const float4* Qg4 = (const float4*)Qb;
    for (int i = tid; i < BLK * DH / 4; i += 128) {
      int row = i >> 4, d4 = (i & 15) << 2;
      float4 q4 = Qg4[i];
      _Float16* dst = sQ + row * LSTR + d4;
      dst[0] = (_Float16)q4.x; dst[1] = (_Float16)q4.y;
      dst[2] = (_Float16)q4.z; dst[3] = (_Float16)q4.w;
    }
  }
  __syncthreads();
  const int  q0  = wq * 16;
  const v16h aQ0 = load_a_frag(sQ, q0 + l16, 0,  half);
  const v16h aQ1 = load_a_frag(sQ, q0 + l16, 32, half);

  v16h vone;
#pragma unroll
  for (int i = 0; i < 16; ++i) vone[i] = (_Float16)1.0f;

  // flash state: O (16x64), row max m, row sum l (l kept as a WMMA accumulator)
  v8f  o[4] = {};
  v8f  lacc = {};
  float mrun[8];
#pragma unroll
  for (int v = 0; v < 8; ++v) mrun[v] = -1e30f;

  for (int it = 0; it < nkb; ++it) {
    const bool apply_tm = (it < rstart);
    const int  kb       = kb_of(it, rb, r0, r1, r2);

#ifdef HAVE_TDM
    const int buf = it & 1;
    if (wq == 0) {
      if (it + 1 < nkb) {   // issue next block's DMA, then wait for current pair
        const int kbn = kb_of(it + 1, rb, r0, r1, r2);
        tdm_load_16kb(Kb + (size_t)kbn * BLK * DH, Kf32[buf ^ 1]);
        tdm_load_16kb(Vb + (size_t)kbn * BLK * DH, Vf32[buf ^ 1]);
        __builtin_amdgcn_s_wait_tensorcnt(2);   // in-order: oldest pair done
      } else {
        __builtin_amdgcn_s_wait_tensorcnt(0);
      }
    }
    __syncthreads();   // f32 staging ready; previous compute done with sKh/sVt

    // LDS f32 -> LDS f16 conversion (+ V transpose)
    {
      const float4* kf4 = (const float4*)Kf32[buf];
      const float4* vf4 = (const float4*)Vf32[buf];
      for (int i = tid; i < BLK * DH / 4; i += 128) {
        int key = i >> 4, d4 = (i & 15) << 2;
        float4 kk = kf4[i];
        float4 vv = vf4[i];
        _Float16* kd = sKh + key * LSTR + d4;
        kd[0] = (_Float16)kk.x; kd[1] = (_Float16)kk.y;
        kd[2] = (_Float16)kk.z; kd[3] = (_Float16)kk.w;
        sVt[(d4 + 0) * LSTR + key] = (_Float16)vv.x;
        sVt[(d4 + 1) * LSTR + key] = (_Float16)vv.y;
        sVt[(d4 + 2) * LSTR + key] = (_Float16)vv.z;
        sVt[(d4 + 3) * LSTR + key] = (_Float16)vv.w;
      }
    }
#else
    __syncthreads();
    {
      const float4* Kg4 = (const float4*)(Kb + (size_t)kb * BLK * DH);
      const float4* Vg4 = (const float4*)(Vb + (size_t)kb * BLK * DH);
      const bool have_next = (it + 1 < nkb);
      const float4* Kg4n = have_next
          ? (const float4*)(Kb + (size_t)kb_of(it + 1, rb, r0, r1, r2) * BLK * DH) : Kg4;
      for (int i = tid; i < BLK * DH / 4; i += 128) {
        int key = i >> 4, d4 = (i & 15) << 2;
        float4 kk = Kg4[i];
        float4 vv = Vg4[i];
        _Float16* kd = sKh + key * LSTR + d4;
        kd[0] = (_Float16)kk.x; kd[1] = (_Float16)kk.y;
        kd[2] = (_Float16)kk.z; kd[3] = (_Float16)kk.w;
        sVt[(d4 + 0) * LSTR + key] = (_Float16)vv.x;
        sVt[(d4 + 1) * LSTR + key] = (_Float16)vv.y;
        sVt[(d4 + 2) * LSTR + key] = (_Float16)vv.z;
        sVt[(d4 + 3) * LSTR + key] = (_Float16)vv.w;
        if (have_next) __builtin_prefetch(&Kg4n[i], 0, 1);
      }
    }
#endif
    __syncthreads();

    // ---- S = Q * K^T ----
    v8f acc[4] = {};
#pragma unroll
    for (int t = 0; t < 4; ++t) {
      v16h bk0 = load_b_frag(sKh, t * 16 + l16, 0, half);
      acc[t] = __builtin_amdgcn_wmma_f32_16x16x32_f16(false, aQ0, false, bk0,
                                                      (short)0, acc[t], false, false);
      v16h bk1 = load_b_frag(sKh, t * 16 + l16, 32, half);
      acc[t] = __builtin_amdgcn_wmma_f32_16x16x32_f16(false, aQ1, false, bk1,
                                                      (short)0, acc[t], false, false);
    }

    // ---- scale + to_mask bias ----
#pragma unroll
    for (int t = 0; t < 4; ++t) {
      float bias = apply_tm ? (1.0f - tmb[kb * BLK + t * 16 + l16]) * NEGV : 0.0f;
#pragma unroll
      for (int v = 0; v < 8; ++v) acc[t][v] = acc[t][v] * 0.125f + bias;
    }

    // ---- online softmax: row max via VALU lane-permute, P = exp(S - m) ----
#pragma unroll
    for (int v = 0; v < 8; ++v) {
      float mx = rowmax16(fmaxf(fmaxf(acc[0][v], acc[1][v]),
                                fmaxf(acc[2][v], acc[3][v])));
      float mnew = fmaxf(mrun[v], mx);
      float corr = __expf(mrun[v] - mnew);
      mrun[v] = mnew;
#pragma unroll
      for (int t = 0; t < 4; ++t) acc[t][v] = __expf(acc[t][v] - mnew);
      o[0][v] *= corr; o[1][v] *= corr; o[2][v] *= corr; o[3][v] *= corr;
      lacc[v] *= corr;
    }

    // ---- P: C-layout -> A-layout via per-wave LDS strip ----
    _Float16* sPw = sP[wq];
#pragma unroll
    for (int t = 0; t < 4; ++t)
#pragma unroll
      for (int v = 0; v < 8; ++v)
        sPw[(half * 8 + v) * LSTR + t * 16 + l16] = (_Float16)acc[t][v];
    asm volatile("s_wait_dscnt 0x0" ::: "memory");
    v16h aP0 = load_a_frag(sPw, l16, 0,  half);
    v16h aP1 = load_a_frag(sPw, l16, 32, half);

    // ---- O += P * V ; row-sum l += P * ones (WMMA, no cross-lane shuffles) ----
    lacc = __builtin_amdgcn_wmma_f32_16x16x32_f16(false, aP0, false, vone,
                                                  (short)0, lacc, false, false);
    lacc = __builtin_amdgcn_wmma_f32_16x16x32_f16(false, aP1, false, vone,
                                                  (short)0, lacc, false, false);
#pragma unroll
    for (int dt = 0; dt < 4; ++dt) {
      v16h bv0 = load_b_frag(sVt, dt * 16 + l16, 0, half);
      o[dt] = __builtin_amdgcn_wmma_f32_16x16x32_f16(false, aP0, false, bv0,
                                                     (short)0, o[dt], false, false);
      v16h bv1 = load_b_frag(sVt, dt * 16 + l16, 32, half);
      o[dt] = __builtin_amdgcn_wmma_f32_16x16x32_f16(false, aP1, false, bv1,
                                                     (short)0, o[dt], false, false);
    }
  }

  // ---- finalize: O /= l ; apply from_mask ; store ----
  float*       ob  = out + (bh * SEQL + (size_t)rb * BLK + q0) * DH;
  const float* fmb = from_mask + (size_t)b * SEQL + (size_t)rb * BLK + q0;
#pragma unroll
  for (int v = 0; v < 8; ++v) {
    int   row = half * 8 + v;
    float fs  = fmb[row] / lacc[v];
#pragma unroll
    for (int dt = 0; dt < 4; ++dt)
      ob[row * DH + dt * 16 + l16] = o[dt][v] * fs;
  }
}

extern "C" void kernel_launch(void* const* d_in, const int* in_sizes, int n_in,
                              void* d_out, int out_size, void* d_ws, size_t ws_size,
                              hipStream_t stream) {
  (void)in_sizes; (void)n_in; (void)out_size; (void)d_ws; (void)ws_size;
  const float* Q         = (const float*)d_in[0];
  const float* K         = (const float*)d_in[1];
  const float* V         = (const float*)d_in[2];
  const int*   rand_attn = (const int*)  d_in[3];
  const float* from_mask = (const float*)d_in[4];
  const float* to_mask   = (const float*)d_in[5];
  float*       out       = (float*)d_out;

  dim3 grid(BSZ * HN * NB);   // one workgroup per (b, h, row-block)
  dim3 block(128);            // 4 waves of 32
  bigbird_sparse_attn_kernel<<<grid, block, 0, stream>>>(
      Q, K, V, rand_attn, from_mask, to_mask, out);
}
